// JordanModel_80487687127278
// MI455X (gfx1250) — matrix operand order repaired and checked by reading, not compiled, not run on hardware
//
#include <hip/hip_runtime.h>

#define INPUT_DIM 512
#define HIDDEN 64
#define OUT_DIM 64
#define W_STRIDE 576            // INPUT_DIM + OUT_DIM (row stride of W_hidden)
#define ROWS_PER_BLOCK 128      // 8 waves * 16 rows
#define THREADS 256

typedef __attribute__((ext_vector_type(16))) __bf16 v16bf;
typedef __attribute__((ext_vector_type(8)))  __bf16 v8bf;
typedef __attribute__((ext_vector_type(8)))  float  v8f;

static __device__ inline v16bf cat8(v8bf lo, v8bf hi) {
  return __builtin_shufflevector(lo, hi, 0,1,2,3,4,5,6,7,8,9,10,11,12,13,14,15);
}

__global__ __launch_bounds__(THREADS) void jordan_fwd(
    const float* __restrict__ x, const float* __restrict__ Wh,
    const float* __restrict__ bh, const float* __restrict__ Wo,
    const float* __restrict__ bo, float* __restrict__ out)
{
  // Weights staged once per workgroup in bf16; per-wave hidden staging tile.
  __shared__ __bf16 sWx[HIDDEN * INPUT_DIM];   // 64 KB
  __shared__ __bf16 sWo[OUT_DIM * HIDDEN];     //  8 KB
  __shared__ __bf16 sHid[8][16 * HIDDEN];      // 16 KB (2 KB per wave)

  const int tid  = threadIdx.x;
  const int wave = tid >> 5;
  const int lane = tid & 31;
  const int half = lane >> 4;   // which 16-lane half of the wave
  const int ln   = lane & 15;

  // ---- stage W_hidden[:, :512] -> LDS bf16 (row-major, stride 512) ----
  for (int i = tid; i < (HIDDEN * INPUT_DIM) / 4; i += THREADS) {
    int h = i >> 7;                 // 128 float4 per row
    int d = (i & 127) << 2;
    float4 w = *(const float4*)(Wh + h * W_STRIDE + d);
    __bf16* p = &sWx[h * INPUT_DIM + d];
    p[0] = (__bf16)w.x; p[1] = (__bf16)w.y; p[2] = (__bf16)w.z; p[3] = (__bf16)w.w;
  }
  // ---- stage W_out -> LDS bf16 (row-major, stride 64) ----
  for (int i = tid; i < (OUT_DIM * HIDDEN) / 4; i += THREADS) {
    int o = i >> 4;                 // 16 float4 per row
    int h = (i & 15) << 2;
    float4 w = *(const float4*)(Wo + o * HIDDEN + h);
    __bf16* p = &sWo[o * HIDDEN + h];
    p[0] = (__bf16)w.x; p[1] = (__bf16)w.y; p[2] = (__bf16)w.z; p[3] = (__bf16)w.w;
  }
  __syncthreads();

  // Per-lane biases: column n = t*16 + ln (same for both lane halves).
  float bhv[4], bov[4];
#pragma unroll
  for (int t = 0; t < 4; ++t) { bhv[t] = bh[t * 16 + ln]; bov[t] = bo[t * 16 + ln]; }

  const int rowBase = blockIdx.x * ROWS_PER_BLOCK + wave * 16;
  const float* xrow = x + (size_t)(rowBase + ln) * INPUT_DIM;

  // ================= GEMM1: hidden_pre = x(16x512) * Wx^T + b_h ============
  // Bias folded into accumulator init: C-layout entry (m, n) == b_h[n],
  // which is constant across a lane's 8 accumulator VGPRs.
  v8f acc[4];
#pragma unroll
  for (int t = 0; t < 4; ++t)
#pragma unroll
    for (int r = 0; r < 8; ++r) acc[t][r] = bhv[t];

  for (int kk = 0; kk < INPUT_DIM / 32; ++kk) {
    const int kb = kk * 32;
    // A tile 16x32 bf16: lane holds row m=ln; K groups {half*8..+7, 16+half*8..+7}
    const float* ap = xrow + kb + half * 8;
    float4 a0 = *(const float4*)(ap);
    float4 a1 = *(const float4*)(ap + 4);
    float4 a2 = *(const float4*)(ap + 16);
    float4 a3 = *(const float4*)(ap + 20);
    __builtin_prefetch(ap + 32, 0, 3);   // next K-slice, near-cache prefetch
    v16bf a;
    a[0]  = (__bf16)a0.x; a[1]  = (__bf16)a0.y; a[2]  = (__bf16)a0.z; a[3]  = (__bf16)a0.w;
    a[4]  = (__bf16)a1.x; a[5]  = (__bf16)a1.y; a[6]  = (__bf16)a1.z; a[7]  = (__bf16)a1.w;
    a[8]  = (__bf16)a2.x; a[9]  = (__bf16)a2.y; a[10] = (__bf16)a2.z; a[11] = (__bf16)a2.w;
    a[12] = (__bf16)a3.x; a[13] = (__bf16)a3.y; a[14] = (__bf16)a3.z; a[15] = (__bf16)a3.w;
#pragma unroll
    for (int t = 0; t < 4; ++t) {
      // B tile 32x16: lane holds col n=t*16+ln; K groups {half*8..+7, 16+half*8..+7}
      const __bf16* bp = &sWx[(t * 16 + ln) * INPUT_DIM + kb + half * 8];
      v16bf b = cat8(*(const v8bf*)bp, *(const v8bf*)(bp + 16));
      acc[t] = __builtin_amdgcn_wmma_f32_16x16x32_bf16(
          false, a, false, b, (short)0, acc[t], false, false);
    }
  }

  // ---- tanh via v_exp + v_rcp, stage 16x64 bf16 hidden tile to LDS ----
  __bf16* hidp = &sHid[wave][0];
#pragma unroll
  for (int t = 0; t < 4; ++t) {
#pragma unroll
    for (int r = 0; r < 8; ++r) {
      float e = __expf(2.0f * acc[t][r]);
      float th = (e - 1.0f) * __builtin_amdgcn_rcpf(e + 1.0f);  // tanh
      int m = r + half * 8;                                     // C-layout row
      hidp[m * HIDDEN + t * 16 + ln] = (__bf16)th;
    }
  }
  // per-wave LDS staging: only need this wave's DS stores complete
  asm volatile("s_wait_dscnt 0x0" ::: "memory");

  // ================= GEMM2: out_pre = hidden(16x64) * Wout^T + b_o =========
  v8f acc2[4];
#pragma unroll
  for (int t = 0; t < 4; ++t)
#pragma unroll
    for (int r = 0; r < 8; ++r) acc2[t][r] = bov[t];

#pragma unroll
  for (int kc = 0; kc < 2; ++kc) {
    const int kb = kc * 32;
    const __bf16* apL = &hidp[ln * HIDDEN + kb + half * 8];
    v16bf a2v = cat8(*(const v8bf*)apL, *(const v8bf*)(apL + 16));
#pragma unroll
    for (int t = 0; t < 4; ++t) {
      const __bf16* bp = &sWo[(t * 16 + ln) * HIDDEN + kb + half * 8];
      v16bf b = cat8(*(const v8bf*)bp, *(const v8bf*)(bp + 16));
      acc2[t] = __builtin_amdgcn_wmma_f32_16x16x32_bf16(
          false, a2v, false, b, (short)0, acc2[t], false, false);
    }
  }

  // ---- sigmoid via v_exp + v_rcp, store f32 output (coalesced) ----
  float* orow = out + (size_t)rowBase * OUT_DIM;
#pragma unroll
  for (int t = 0; t < 4; ++t) {
#pragma unroll
    for (int r = 0; r < 8; ++r) {
      float s = __builtin_amdgcn_rcpf(1.0f + __expf(-acc2[t][r]));
      int m = r + half * 8;
      orow[m * OUT_DIM + t * 16 + ln] = s;
    }
  }
}

extern "C" void kernel_launch(void* const* d_in, const int* in_sizes, int n_in,
                              void* d_out, int out_size, void* d_ws, size_t ws_size,
                              hipStream_t stream) {
  const float* x  = (const float*)d_in[0];
  const float* Wh = (const float*)d_in[1];
  const float* bh = (const float*)d_in[2];
  const float* Wo = (const float*)d_in[3];
  const float* bo = (const float*)d_in[4];
  float* out = (float*)d_out;

  const int batch = in_sizes[0] / INPUT_DIM;      // 262144
  const int grid  = batch / ROWS_PER_BLOCK;       // 2048
  jordan_fwd<<<grid, THREADS, 0, stream>>>(x, Wh, bh, Wo, bo, out);
}
